// QWenAttention_62749472195077
// MI455X (gfx1250) — matrix operand order repaired and checked by reading, not compiled, not run on hardware
//
#include <hip/hip_runtime.h>
#include <hip/hip_bf16.h>

#define S_LEN 2048
#define HIDDEN_DIM 4096
#define NHEADS 32
#define HDIM 128
#define QKV_N (3 * HIDDEN_DIM)

typedef unsigned short u16;
typedef __bf16 bf16x16 __attribute__((ext_vector_type(16)));
typedef float  floatx8 __attribute__((ext_vector_type(8)));

// ---------- helpers ----------

static __device__ __forceinline__ u16 f2bf(float f) {
  unsigned int u = __float_as_uint(f);
  u += 0x7FFFu + ((u >> 16) & 1u);   // round-to-nearest-even
  return (u16)(u >> 16);
}

struct Pair16 { uint4 a, b; };

// Build a 16-element bf16 fragment from two 16-byte chunks (lo at p, hi at p+hiElems).
static __device__ __forceinline__ bf16x16 frag(const u16* p, int hiElems) {
  Pair16 t;
  t.a = *(const uint4*)(p);
  t.b = *(const uint4*)(p + hiElems);
  return __builtin_bit_cast(bf16x16, t);
}

static __device__ __forceinline__ floatx8 wmma_bf16(bf16x16 a, bf16x16 b, floatx8 c) {
  // D = A(16x32 bf16) * B(32x16 bf16) + C(16x16 f32)
  return __builtin_amdgcn_wmma_f32_16x16x32_bf16(
      /*neg_a=*/false, a, /*neg_b=*/false, b,
      /*c_mod=*/(short)0, c, /*reuse_a=*/false, /*reuse_b=*/false);
}

static __device__ __forceinline__ floatx8 zero8() {
  floatx8 z;
#pragma unroll
  for (int i = 0; i < 8; ++i) z[i] = 0.0f;
  return z;
}

// Async memory->LDS copy of 8 bytes per lane (ASYNCcnt path; no VGPR data round-trip).
static __device__ __forceinline__ void async_copy_b64(const u16* gsrc, u16* ldst) {
  const unsigned lds = (unsigned)(size_t)ldst;   // generic LDS addr: low 32 bits = LDS offset
  asm volatile("global_load_async_to_lds_b64 %0, %1, off"
               :: "v"(lds), "v"(gsrc) : "memory");
}

static __device__ __forceinline__ void wait_async0() {
  asm volatile("s_wait_asynccnt 0x0" ::: "memory");
}

// ---------- kernel 1 & 4: row-major GEMM, C = A*B (+bias), bf16 WMMA, f32 accumulate ----------
// A: [M][K] (fp32 or bf16), B: [K][N] fp32 (converted to bf16 on staging), C: [M][N] fp32.
// Block tile 128x128, K-step 32; 8 waves, each wave computes a 32x64 region (2x4 WMMA tiles).
// bf16-A path stages its tile with global_load_async_to_lds_b64 (pure copy, no conversion).

template <bool A_BF16, bool HAS_BIAS>
__global__ void __launch_bounds__(256) gemm_rowmajor_bf16(
    const void* __restrict__ Ain, const float* __restrict__ B,
    const float* __restrict__ bias, float* __restrict__ C,
    const int M, const int N, const int K) {
  __shared__ u16 As[128 * 40];  // [row 0..127][k 0..31], stride 40 (80B = 5*16B)
  __shared__ u16 Bs[128 * 40];  // transposed: [n 0..127][k 0..31], stride 40

  const int tid  = threadIdx.x;
  const int wid  = tid >> 5;
  const int lane = tid & 31;
  const int wm = wid >> 1;        // 0..3  -> 32-row band
  const int wn = wid & 1;         // 0..1  -> 64-col band
  const int lm = lane & 15;
  const int lg = lane >> 4;
  const int tileM = blockIdx.y * 128;
  const int tileN = blockIdx.x * 128;

  floatx8 acc[2][4];
#pragma unroll
  for (int i = 0; i < 2; ++i)
#pragma unroll
    for (int j = 0; j < 4; ++j) acc[i][j] = zero8();

  const int tr = tid >> 3;            // A loader: row within 32-row pass
  const int tc = (tid & 7) * 4;       // A loader: k column (x4)
  const int bk = tid >> 5;            // B loader: k row within 8-row pass
  const int bn = (tid & 31) * 4;      // B loader: n column (x4)

  for (int k0 = 0; k0 < K; k0 += 32) {
    __syncthreads();
    // --- stage A tile (128 x 32) -> bf16 ---
    if (A_BF16) {
      const u16* A = (const u16*)Ain;
#pragma unroll
      for (int p = 0; p < 4; ++p) {
        const int row = p * 32 + tr;
        async_copy_b64(&A[(size_t)(tileM + row) * K + k0 + tc], &As[row * 40 + tc]);
      }
    } else {
      const float* A = (const float*)Ain;
#pragma unroll
      for (int p = 0; p < 4; ++p) {
        const int row = p * 32 + tr;
        float4 v = *(const float4*)&A[(size_t)(tileM + row) * K + k0 + tc];
        uint2 pk;
        pk.x = (unsigned)f2bf(v.x) | ((unsigned)f2bf(v.y) << 16);
        pk.y = (unsigned)f2bf(v.z) | ((unsigned)f2bf(v.w) << 16);
        *(uint2*)&As[row * 40 + tc] = pk;
      }
    }
    // --- stage B tile (32 x 128), transposed into LDS as [n][k] ---
#pragma unroll
    for (int p = 0; p < 4; ++p) {
      const int kr = p * 8 + bk;
      float4 v = *(const float4*)&B[(size_t)(k0 + kr) * N + tileN + bn];
      Bs[(bn + 0) * 40 + kr] = f2bf(v.x);
      Bs[(bn + 1) * 40 + kr] = f2bf(v.y);
      Bs[(bn + 2) * 40 + kr] = f2bf(v.z);
      Bs[(bn + 3) * 40 + kr] = f2bf(v.w);
    }
    if (k0 + 32 < K) {  // prefetch next B tile line -> global_prefetch_b8
      __builtin_prefetch(&B[(size_t)(k0 + 32 + bk) * N + tileN + bn], 0, 0);
    }
    if (A_BF16) wait_async0();   // drain ASYNCcnt before the barrier (barrier doesn't)
    __syncthreads();

    // --- compute: 2x4 WMMA tiles per wave ---
    bf16x16 af[2];
#pragma unroll
    for (int tm = 0; tm < 2; ++tm) {
      // A layout: lane m=lm, K quarters {lg*8..+7} and {16+lg*8..+7}
      const u16* p = &As[(wm * 32 + tm * 16 + lm) * 40 + lg * 8];
      af[tm] = frag(p, 16);
    }
#pragma unroll
    for (int tn = 0; tn < 4; ++tn) {
      // B layout: lane n=lm, K quarters {lg*8..+7} and {16+lg*8..+7}
      const u16* p = &Bs[(wn * 64 + tn * 16 + lm) * 40 + lg * 8];
      bf16x16 bfr = frag(p, 16);
#pragma unroll
      for (int tm = 0; tm < 2; ++tm)
        acc[tm][tn] = wmma_bf16(af[tm], bfr, acc[tm][tn]);
    }
  }

  // --- epilogue: D layout rows m = lg*8 + i, col n = lm ---
#pragma unroll
  for (int tm = 0; tm < 2; ++tm) {
#pragma unroll
    for (int tn = 0; tn < 4; ++tn) {
      const int gm0 = tileM + wm * 32 + tm * 16 + lg * 8;
      const int gn  = tileN + wn * 64 + tn * 16 + lm;
      const float bb = HAS_BIAS ? bias[gn] : 0.0f;
#pragma unroll
      for (int i = 0; i < 8; ++i)
        C[(size_t)(gm0 + i) * N + gn] = acc[tm][tn][i] + bb;
    }
  }
}

// ---------- kernel 2: RoPE + per-head bf16 scatter ----------
// qkv fp32 [S][3*H] -> Qh/Kh/Vh bf16 [head][S][128] (Q,K rotated; V copied)

__global__ void __launch_bounds__(256) rope_scatter(
    const int* __restrict__ pos, const float* __restrict__ qkv,
    u16* __restrict__ Qh, u16* __restrict__ Kh, u16* __restrict__ Vh) {
  const int idx = blockIdx.x * 256 + threadIdx.x;   // (s, h, d-pair)
  if (idx >= S_LEN * NHEADS * 64) return;
  const int d = idx & 63;
  const int h = (idx >> 6) & 31;
  const int s = idx >> 11;

  const float p   = (float)pos[s];
  const float inv = __powf(10000.0f, -(float)d * (1.0f / 64.0f));
  float sn, cs;
  __sincosf(p * inv, &sn, &cs);

  const float* base = qkv + (size_t)s * QKV_N + h * HDIM;
  const float q1 = base[d],                  q2 = base[64 + d];
  const float k1 = base[HIDDEN_DIM + d],     k2 = base[HIDDEN_DIM + 64 + d];
  const float v1 = base[2 * HIDDEN_DIM + d], v2 = base[2 * HIDDEN_DIM + 64 + d];

  const size_t o = ((size_t)h * S_LEN + s) * HDIM;
  Qh[o + d]      = f2bf(q1 * cs - q2 * sn);
  Qh[o + 64 + d] = f2bf(q2 * cs + q1 * sn);
  Kh[o + d]      = f2bf(k1 * cs - k2 * sn);
  Kh[o + 64 + d] = f2bf(k2 * cs + k1 * sn);
  Vh[o + d]      = f2bf(v1);
  Vh[o + 64 + d] = f2bf(v2);
}

// ---------- kernel 3: causal flash attention, one (head, 128-row Q block) per workgroup ----------
// Br=128 (16 rows/wave x 8 waves), Bc=64. Q fragments kept in registers across the K loop.
// K block staged with async memory->LDS copies (ASYNCcnt); V transposed through VGPRs.

__global__ void __launch_bounds__(256) flash_attn(
    const u16* __restrict__ Qh, const u16* __restrict__ Kh,
    const u16* __restrict__ Vh, u16* __restrict__ Attn) {
  __shared__ u16 Ks[64 * 136];      // [key 0..63][d 0..127], stride 136 (272B = 17*16B)
  __shared__ u16 Vs[128 * 72];      // transposed: [d 0..127][key 0..63], stride 72
  __shared__ u16 Ps[8][16 * 72];    // per-wave P block [m 0..15][key 0..63], stride 72

  const int h  = blockIdx.y;
  const int qb = blockIdx.x;
  const int tid  = threadIdx.x;
  const int wid  = tid >> 5;
  const int lane = tid & 31;
  const int lm = lane & 15;
  const int lg = lane >> 4;
  const int qrow0 = qb * 128 + wid * 16;

  // Q fragments: 16 rows x 128 dims, 4 K-steps of 32
  bf16x16 qf[4];
  {
    const u16* qbase = Qh + ((size_t)h * S_LEN + qrow0 + lm) * HDIM;
#pragma unroll
    for (int kk = 0; kk < 4; ++kk)
      qf[kk] = frag(qbase + kk * 32 + lg * 8, 16);
  }

  floatx8 oacc[8];
#pragma unroll
  for (int i = 0; i < 8; ++i) oacc[i] = zero8();
  float mrow[8], lrow[8];
#pragma unroll
  for (int i = 0; i < 8; ++i) { mrow[i] = -1.0e30f; lrow[i] = 0.0f; }

  const float cs = 0.08838834764831845f * 1.4426950408889634f; // rsqrt(128) * log2(e)

  const int kbMax = 2 * qb + 1;
  for (int kb = 0; kb <= kbMax; ++kb) {
    __syncthreads();
    // --- stage K (async row-major copy) and V (transposed via VGPRs): 64 x 128 bf16 each ---
#pragma unroll
    for (int p = 0; p < 8; ++p) {
      const int idx = p * 256 + tid;       // 0..2047 quads
      const int r = idx >> 5;              // key row 0..63
      const int c = (idx & 31) * 4;        // dim col
      const size_t src = ((size_t)h * S_LEN + kb * 64 + r) * HDIM + c;
      async_copy_b64(&Kh[src], &Ks[r * 136 + c]);   // memory -> LDS, no VGPR data
      const u16* vp = &Vh[src];
      Vs[(c + 0) * 72 + r] = vp[0];
      Vs[(c + 1) * 72 + r] = vp[1];
      Vs[(c + 2) * 72 + r] = vp[2];
      Vs[(c + 3) * 72 + r] = vp[3];
    }
    wait_async0();                          // drain ASYNCcnt before the barrier
    __syncthreads();

    // --- S = Q * K^T  (16 x 64 per wave; 4 n-tiles x 4 k-steps) ---
    floatx8 sacc[4];
#pragma unroll
    for (int tn = 0; tn < 4; ++tn) sacc[tn] = zero8();
#pragma unroll
    for (int kk = 0; kk < 4; ++kk) {
#pragma unroll
      for (int tn = 0; tn < 4; ++tn) {
        bf16x16 bfr = frag(&Ks[(tn * 16 + lm) * 136 + kk * 32 + lg * 8], 16);
        sacc[tn] = wmma_bf16(qf[kk], bfr, sacc[tn]);
      }
    }

    // --- causal mask (only near the diagonal) ---
    if (kb * 64 + 63 > qrow0) {
#pragma unroll
      for (int tn = 0; tn < 4; ++tn) {
        const int key = kb * 64 + tn * 16 + lm;
#pragma unroll
        for (int i = 0; i < 8; ++i) {
          const int qr = qrow0 + lg * 8 + i;
          if (key > qr) sacc[tn][i] = -3.0e38f;
        }
      }
    }

    // --- online softmax (per row m = lg*8 + i; reduce across 16 lanes of each half) ---
    float sf[8];
#pragma unroll
    for (int i = 0; i < 8; ++i) {
      float mx = fmaxf(fmaxf(sacc[0][i], sacc[1][i]), fmaxf(sacc[2][i], sacc[3][i]));
      mx *= cs;
#pragma unroll
      for (int off = 8; off > 0; off >>= 1)
        mx = fmaxf(mx, __shfl_xor(mx, off, 32));
      const float mnew = fmaxf(mrow[i], mx);
      sf[i] = exp2f(mrow[i] - mnew);
      mrow[i] = mnew;
      float rs = 0.0f;
#pragma unroll
      for (int tn = 0; tn < 4; ++tn) {
        const float pv = exp2f(sacc[tn][i] * cs - mnew);
        sacc[tn][i] = pv;
        rs += pv;
      }
#pragma unroll
      for (int off = 8; off > 0; off >>= 1)
        rs += __shfl_xor(rs, off, 32);
      lrow[i] = lrow[i] * sf[i] + rs;
    }
#pragma unroll
    for (int tn = 0; tn < 8; ++tn)
#pragma unroll
      for (int i = 0; i < 8; ++i) oacc[tn][i] *= sf[i];

    // --- P -> LDS (bf16), cross-lane reshuffle for A-operand layout ---
#pragma unroll
    for (int tn = 0; tn < 4; ++tn)
#pragma unroll
      for (int i = 0; i < 8; ++i)
        Ps[wid][(lg * 8 + i) * 72 + tn * 16 + lm] = f2bf(sacc[tn][i]);
    // Wave-private LDS round-trip: drain DS ops before fragment reload.
    asm volatile("s_wait_dscnt 0" ::: "memory");

    // --- O += P * V  (16 x 128 per wave; 8 n-tiles x 2 k-steps of 32 keys) ---
#pragma unroll
    for (int kk = 0; kk < 2; ++kk) {
      bf16x16 pf = frag(&Ps[wid][lm * 72 + kk * 32 + lg * 8], 16);
#pragma unroll
      for (int tn = 0; tn < 8; ++tn) {
        bf16x16 vf = frag(&Vs[(tn * 16 + lm) * 72 + kk * 32 + lg * 8], 16);
        oacc[tn] = wmma_bf16(pf, vf, oacc[tn]);
      }
    }
  }

  // --- epilogue: normalize, store bf16 to attn buffer [s][h*128 + d] ---
#pragma unroll
  for (int tn = 0; tn < 8; ++tn) {
#pragma unroll
    for (int i = 0; i < 8; ++i) {
      const float o = oacc[tn][i] / lrow[i];
      const int s = qrow0 + lg * 8 + i;
      Attn[(size_t)s * HIDDEN_DIM + h * HDIM + tn * 16 + lm] = f2bf(o);
    }
  }
}

// ---------- launch ----------

extern "C" void kernel_launch(void* const* d_in, const int* in_sizes, int n_in,
                              void* d_out, int out_size, void* d_ws, size_t ws_size,
                              hipStream_t stream) {
  (void)in_sizes; (void)n_in; (void)out_size; (void)ws_size;
  const int*   positions = (const int*)d_in[0];
  const float* hidden    = (const float*)d_in[1];
  const float* w_qkv     = (const float*)d_in[2];
  const float* b_qkv     = (const float*)d_in[3];
  const float* w_o       = (const float*)d_in[4];
  float* out = (float*)d_out;

  // workspace layout: qkv fp32 (100.7MB) | Qh | Kh | Vh | Attn (bf16, 16.8MB each)
  float* qkv = (float*)d_ws;
  u16* Qh   = (u16*)(qkv + (size_t)S_LEN * QKV_N);
  u16* Kh   = Qh + (size_t)NHEADS * S_LEN * HDIM;
  u16* Vh   = Kh + (size_t)NHEADS * S_LEN * HDIM;
  u16* Attn = Vh + (size_t)NHEADS * S_LEN * HDIM;

  // 1) qkv = hidden @ w_qkv + b_qkv
  gemm_rowmajor_bf16<false, true>
      <<<dim3(QKV_N / 128, S_LEN / 128), 256, 0, stream>>>(
          hidden, w_qkv, b_qkv, qkv, S_LEN, QKV_N, HIDDEN_DIM);

  // 2) RoPE + per-head bf16 layout
  rope_scatter<<<(S_LEN * NHEADS * 64) / 256, 256, 0, stream>>>(
      positions, qkv, Qh, Kh, Vh);

  // 3) causal flash attention
  flash_attn<<<dim3(S_LEN / 128, NHEADS), 256, 0, stream>>>(Qh, Kh, Vh, Attn);

  // 4) out = attn @ w_o
  gemm_rowmajor_bf16<true, false>
      <<<dim3(HIDDEN_DIM / 128, S_LEN / 128), 256, 0, stream>>>(
          Attn, w_o, nullptr, out, S_LEN, HIDDEN_DIM, HIDDEN_DIM);
}
